// MultiHeadAttention_60120952210042
// MI455X (gfx1250) — compile-verified, matrix operand order checked by reading, not run on hardware
//
#include <hip/hip_runtime.h>

// ---------------------------------------------------------------------------
// MultiHeadAttention for MI455X (gfx1250, wave32, WMMA 16x16x32 f16)
//
// N=8, T_Q=T_K=2048, d_model=512, H=8, head_dim=64.
// Kernel 1: Q/K/V projections (f32 in -> f16 workspace), V stored transposed.
// Kernel 2: flash attention, fp32 online softmax, WMMA for S=QK^T and O=P V,
//           double-buffered GLOBAL_LOAD_ASYNC_TO_LDS_B128 staging (ASYNCcnt).
// Workspace: 3 * 8*2048*512 f16 = 48 MB  (Q | K | V^T)
// ---------------------------------------------------------------------------

typedef __attribute__((ext_vector_type(16))) _Float16 v16h;
typedef __attribute__((ext_vector_type(8)))  float    v8f;
typedef __attribute__((ext_vector_type(4)))  _Float16 v4h;

union F16x16 { v16h v; uint4 q[2]; };

__device__ __forceinline__ v8f wmma_f16(const v16h& a, const v16h& b, const v8f& c) {
  return __builtin_amdgcn_wmma_f32_16x16x32_f16(false, a, false, b, (short)0, c, false, false);
}

// async copy 16B global -> LDS, tracked by ASYNCcnt (per-lane addresses)
__device__ __forceinline__ void async_g2l_b128(const _Float16* gptr, void* lptr) {
  const unsigned lds_off = (unsigned)(unsigned long long)lptr;       // low 32b of flat addr == LDS offset
  const unsigned long long gaddr = (unsigned long long)gptr;
  asm volatile("global_load_async_to_lds_b128 %0, %1, off"
               :: "v"(lds_off), "v"(gaddr) : "memory");
}

#define NB      8
#define TQ      2048
#define DMODEL  512
#define NHEADS  8
#define HDIM    64
#define ELEMS   ((size_t)NB * TQ * DMODEL)   // 8,388,608 halves per tensor

// ============================ Kernel 1: projections =========================
// grid: (16384/128, 512/128, 3), block 256.  which: 0=Q, 1=K, 2=V(transposed)
__global__ __launch_bounds__(256) void proj_gemm_kernel(
    const float* __restrict__ x, const float* __restrict__ key,
    const float* __restrict__ Wq, const float* __restrict__ Wk,
    const float* __restrict__ Wv, _Float16* __restrict__ ws)
{
  const int which = blockIdx.z;
  const float* __restrict__ src = (which == 0) ? x : key;
  const float* __restrict__ W   = (which == 0) ? Wq : ((which == 1) ? Wk : Wv);
  _Float16* __restrict__ dst    = ws + (size_t)which * ELEMS;

  __shared__ _Float16 lA [128][40];   // X tile  [m][k], +8 pad
  __shared__ _Float16 lBt[128][40];   // W tile transposed [n][k], +8 pad

  const int tid  = threadIdx.x;
  const int lane = tid & 31;
  const int wv   = tid >> 5;          // 8 waves
  const int wm   = wv & 1;            // 2 x 64 rows
  const int wn   = wv >> 1;           // 4 x 32 cols
  const int l16  = lane & 15;
  const int lh   = lane >> 4;         // half-wave select for K-runs

  const int mbase = blockIdx.x * 128; // over N*T = 16384 rows
  const int nbase = blockIdx.y * 128; // over 512 cols

  v8f acc[4][2];
#pragma unroll
  for (int mi = 0; mi < 4; ++mi)
#pragma unroll
    for (int ni = 0; ni < 2; ++ni) acc[mi][ni] = {};

  for (int kb = 0; kb < DMODEL / 32; ++kb) {
    const int k0 = kb * 32;
    __syncthreads();
    // stage A tile 128x32 (f32 -> f16)
#pragma unroll
    for (int i = 0; i < 4; ++i) {
      const int c = tid + i * 256;            // 0..1023 float4 chunks
      const int row = c >> 3, col4 = c & 7;
      const float4 f = *(const float4*)(src + (size_t)(mbase + row) * DMODEL + k0 + col4 * 4);
      v4h h; h[0] = (_Float16)f.x; h[1] = (_Float16)f.y; h[2] = (_Float16)f.z; h[3] = (_Float16)f.w;
      *(v4h*)&lA[row][col4 * 4] = h;
    }
    // stage W tile 32x128, transposed into lBt[n][k]
#pragma unroll
    for (int i = 0; i < 4; ++i) {
      const int c = tid + i * 256;
      const int krow = c >> 5, nc4 = c & 31;
      const float4 f = *(const float4*)(W + (size_t)(k0 + krow) * DMODEL + nbase + nc4 * 4);
      lBt[nc4 * 4 + 0][krow] = (_Float16)f.x;
      lBt[nc4 * 4 + 1][krow] = (_Float16)f.y;
      lBt[nc4 * 4 + 2][krow] = (_Float16)f.z;
      lBt[nc4 * 4 + 3][krow] = (_Float16)f.w;
    }
    __syncthreads();

    F16x16 af[4], bf[2];
#pragma unroll
    for (int mi = 0; mi < 4; ++mi) {
      const _Float16* p = &lA[wm * 64 + mi * 16 + l16][lh * 8];
      af[mi].q[0] = *(const uint4*)p;
      af[mi].q[1] = *(const uint4*)(p + 16);
    }
#pragma unroll
    for (int ni = 0; ni < 2; ++ni) {
      const _Float16* p = &lBt[wn * 32 + ni * 16 + l16][lh * 8];
      bf[ni].q[0] = *(const uint4*)p;
      bf[ni].q[1] = *(const uint4*)(p + 16);
    }
#pragma unroll
    for (int mi = 0; mi < 4; ++mi)
#pragma unroll
      for (int ni = 0; ni < 2; ++ni)
        acc[mi][ni] = wmma_f16(af[mi].v, bf[ni].v, acc[mi][ni]);
  }

  // store: Q/K row-major [N*T][512]; V transposed [N][512][T]
#pragma unroll
  for (int mi = 0; mi < 4; ++mi)
#pragma unroll
    for (int ni = 0; ni < 2; ++ni)
#pragma unroll
      for (int r = 0; r < 8; ++r) {
        const int mg = mbase + wm * 64 + mi * 16 + r + lh * 8;
        const int ug = nbase + wn * 32 + ni * 16 + l16;
        const _Float16 hv = (_Float16)acc[mi][ni][r];
        if (which < 2) {
          dst[(size_t)mg * DMODEL + ug] = hv;
        } else {
          const int nb = mg >> 11;          // batch
          const int t  = mg & 2047;         // time
          dst[((size_t)nb * DMODEL + ug) * TQ + t] = hv;
        }
      }
}

// ============================ Kernel 2: flash attention =====================
// grid: (T_Q/128, H, N), block 256 (8 waves, 16 q-rows each). 64-key tiles,
// double-buffered async global->LDS staging.
__global__ __launch_bounds__(256) void flash_attn_kernel(
    const _Float16* __restrict__ ws, float* __restrict__ out)
{
  const _Float16* __restrict__ Qh = ws;
  const _Float16* __restrict__ Kh = ws + ELEMS;
  const _Float16* __restrict__ Vt = ws + 2 * ELEMS;

  const int qb = blockIdx.x;   // q block (128 rows)
  const int h  = blockIdx.y;
  const int n  = blockIdx.z;

  __shared__ _Float16 lK[2][64][72];     // K tile  [tk][d], +8 pad, 2 buffers
  __shared__ _Float16 lV[2][64][72];     // V^T tile [d][tk], +8 pad, 2 buffers
  __shared__ _Float16 lP[8][16][72];     // per-wave probs [q][tk], +8 pad

  const int tid = threadIdx.x, lane = tid & 31, wv = tid >> 5;
  const int l16 = lane & 15, lh = lane >> 4;

  // per-thread staging coordinates: 2 x 16B chunks of K and of V^T per tile
  const int c0row = tid >> 3,          c0c8 = tid & 7;          // chunks 0..255
  const int c1row = (tid + 256) >> 3,  c1c8 = (tid + 256) & 7;  // chunks 256..511

  // issue the 4 async copies for tile kt into buffer buf
  auto stage_tile = [&](int kt, int buf) {
    const int tk0 = kt * 64;
    async_g2l_b128(Kh + ((size_t)(n * TQ + tk0 + c0row) * DMODEL + h * HDIM + c0c8 * 8),
                   &lK[buf][c0row][c0c8 * 8]);
    async_g2l_b128(Kh + ((size_t)(n * TQ + tk0 + c1row) * DMODEL + h * HDIM + c1c8 * 8),
                   &lK[buf][c1row][c1c8 * 8]);
    async_g2l_b128(Vt + ((size_t)(n * DMODEL + h * HDIM + c0row) * TQ + tk0 + c0c8 * 8),
                   &lV[buf][c0row][c0c8 * 8]);
    async_g2l_b128(Vt + ((size_t)(n * DMODEL + h * HDIM + c1row) * TQ + tk0 + c1c8 * 8),
                   &lV[buf][c1row][c1c8 * 8]);
  };

  // Q A-fragments (16 rows x 64 d), loaded once, reused over all K tiles
  F16x16 aq[2];
  {
    const int qrow = qb * 128 + wv * 16 + l16;
    const _Float16* base = Qh + ((size_t)(n * TQ + qrow) * DMODEL + h * HDIM);
#pragma unroll
    for (int kk = 0; kk < 2; ++kk) {
      const _Float16* p = base + kk * 32 + lh * 8;
      aq[kk].q[0] = *(const uint4*)p;
      aq[kk].q[1] = *(const uint4*)(p + 16);
    }
  }

  v8f o[4];
#pragma unroll
  for (int dt = 0; dt < 4; ++dt) o[dt] = {};
  float mrow[8], lrow[8];
#pragma unroll
  for (int r = 0; r < 8; ++r) { mrow[r] = -__builtin_inff(); lrow[r] = 0.0f; }

  const float SCALE = 0.044194173824159216f;   // 1/sqrt(KEY_DIM=512)
  const int NT = TQ / 64;                      // 32 key tiles

  stage_tile(0, 0);                            // prime the pipeline

  for (int kt = 0; kt < NT; ++kt) {
    const int buf = kt & 1;
    __syncthreads();                            // prev compute done; other buffer free
    if (kt + 1 < NT) {
      stage_tile(kt + 1, buf ^ 1);              // overlap next tile's copy with this compute
      asm volatile("s_wait_asynccnt 0x4" ::: "memory");  // tile kt's 4 ops complete (in-order)
    } else {
      asm volatile("s_wait_asynccnt 0x0" ::: "memory");
    }
    __syncthreads();                            // publish LDS tile to all waves

    // S = Q K^T  (4 tiles of 16x16, K-dim 64)
    v8f s[4];
#pragma unroll
    for (int t = 0; t < 4; ++t) {
      s[t] = {};
#pragma unroll
      for (int kk = 0; kk < 2; ++kk) {
        F16x16 bk;
        const _Float16* p = &lK[buf][t * 16 + l16][kk * 32 + lh * 8];
        bk.q[0] = *(const uint4*)p;
        bk.q[1] = *(const uint4*)(p + 16);
        s[t] = wmma_f16(aq[kk].v, bk.v, s[t]);
      }
      s[t] = s[t] * SCALE;
    }

    // online softmax; rows live in one 16-lane half -> shfl_xor masks 1..8
    float pv[4][8];
#pragma unroll
    for (int r = 0; r < 8; ++r) {
      float v = fmaxf(fmaxf(s[0][r], s[1][r]), fmaxf(s[2][r], s[3][r]));
#pragma unroll
      for (int off = 1; off < 16; off <<= 1) v = fmaxf(v, __shfl_xor(v, off, 32));
      const float mnew  = fmaxf(mrow[r], v);
      const float alpha = __expf(mrow[r] - mnew);
      float rs = 0.0f;
#pragma unroll
      for (int t = 0; t < 4; ++t) { const float e = __expf(s[t][r] - mnew); pv[t][r] = e; rs += e; }
#pragma unroll
      for (int off = 1; off < 16; off <<= 1) rs += __shfl_xor(rs, off, 32);
      lrow[r] = lrow[r] * alpha + rs;
      mrow[r] = mnew;
#pragma unroll
      for (int dt = 0; dt < 4; ++dt) o[dt][r] *= alpha;
    }

    // C-layout -> A-layout via per-wave LDS staging of P (f16)
#pragma unroll
    for (int t = 0; t < 4; ++t)
#pragma unroll
      for (int r = 0; r < 8; ++r)
        lP[wv][r + lh * 8][t * 16 + l16] = (_Float16)pv[t][r];
    asm volatile("s_wait_dscnt 0" ::: "memory");   // wave-local LDS visibility

    // O += P V   (A = P 16x64, B = V 64x16 per d-tile)
#pragma unroll
    for (int kk = 0; kk < 2; ++kk) {
      F16x16 ap;
      const _Float16* pp = &lP[wv][l16][kk * 32 + lh * 8];
      ap.q[0] = *(const uint4*)pp;
      ap.q[1] = *(const uint4*)(pp + 16);
#pragma unroll
      for (int dt = 0; dt < 4; ++dt) {
        F16x16 bv;
        const _Float16* pb = &lV[buf][dt * 16 + l16][kk * 32 + lh * 8];
        bv.q[0] = *(const uint4*)pb;
        bv.q[1] = *(const uint4*)(pb + 16);
        o[dt] = wmma_f16(ap.v, bv.v, o[dt]);
      }
    }
  }

  // normalize and write fp32 output [N][T][512]
#pragma unroll
  for (int r = 0; r < 8; ++r) {
    const float inv = 1.0f / lrow[r];
    const int qg = qb * 128 + wv * 16 + r + lh * 8;
    float* dst = out + (size_t)(n * TQ + qg) * DMODEL + h * HDIM;
#pragma unroll
    for (int dt = 0; dt < 4; ++dt)
      dst[dt * 16 + l16] = o[dt][r] * inv;
  }
}

// ================================ launcher ==================================
extern "C" void kernel_launch(void* const* d_in, const int* in_sizes, int n_in,
                              void* d_out, int out_size, void* d_ws, size_t ws_size,
                              hipStream_t stream) {
  (void)in_sizes; (void)n_in; (void)out_size; (void)ws_size;
  const float* x   = (const float*)d_in[0];
  const float* key = (const float*)d_in[1];
  const float* Wq  = (const float*)d_in[2];
  const float* Wk  = (const float*)d_in[3];
  const float* Wv  = (const float*)d_in[4];
  _Float16* ws = (_Float16*)d_ws;      // needs 3*8*2048*512*2 = 48 MB
  float* out = (float*)d_out;

  dim3 g1((NB * TQ) / 128, DMODEL / 128, 3);
  proj_gemm_kernel<<<g1, 256, 0, stream>>>(x, key, Wq, Wk, Wv, ws);

  dim3 g2(TQ / 128, NHEADS, NB);
  flash_attn_kernel<<<g2, 256, 0, stream>>>(ws, out);
}
